// CopyTokenDecoder_89653147337412
// MI455X (gfx1250) — compile-verified, matrix operand order checked by reading, not run on hardware
//
#include <hip/hip_runtime.h>

// ---------------------------------------------------------------------------
// CopyTokenDecoder on MI455X (gfx1250).
// Heavy GEMMs: bf16 WMMA (v_wmma_f32_16x16x32_bf16), f32 accumulation,
// 128x128 C-tiles per 256-thread (8-wave, wave32) workgroup. A/B tiles are
// DMA'd into double-buffered LDS by the Tensor Data Mover
// (tensor_load_to_lds + s_wait_tensorcnt), with TDM's native LDS padding
// reproducing the bank-conflict-avoiding 40-element row stride.
// Vocab logits go straight into d_out; softmax+gates+copy-scatter in place;
// final log pass is a separate kernel (global-atomics visibility).
// ---------------------------------------------------------------------------

typedef __attribute__((ext_vector_type(16))) __bf16 v16bf;
typedef __attribute__((ext_vector_type(8)))  __bf16 v8bf;
typedef __attribute__((ext_vector_type(8)))  float  v8f;
typedef __attribute__((ext_vector_type(4)))  unsigned int u32x4;
typedef __attribute__((ext_vector_type(8)))  int  i32x8;
typedef __attribute__((ext_vector_type(4)))  int  i32x4;

#define V_   32000
#define E_   1024
#define FF_  4096
#define T_   256
#define B_   8
#define S_   512
#define LN_EPS 1e-5f

__device__ __forceinline__ __bf16 f2bf(float f) {
  unsigned u = __float_as_uint(f);
  u += 0x7fffu + ((u >> 16) & 1u);          // round-to-nearest-even
  unsigned short h = (unsigned short)(u >> 16);
  return __builtin_bit_cast(__bf16, h);
}

// ------------------------------- fp32 -> bf16 ------------------------------
__global__ void cvt_bf16_kernel(const float* __restrict__ in,
                                __bf16* __restrict__ out, long n) {
  long i = (long)blockIdx.x * blockDim.x + threadIdx.x;
  long stride = (long)gridDim.x * blockDim.x;
  for (; i < n; i += stride) out[i] = f2bf(in[i]);
}

// ------------------------------ WMMA GEMM ----------------------------------
// C[M,N] = act( (A[M,K] @ W[N,K]^T + bias) * alpha ), A/W bf16, C f32 and/or
// bf16 mirror. Batched via blockIdx.z with element strides.
#define BM 128
#define BN 128
#define BK 32
#define LDT 40                 // padded LDS K-stride (80B rows)
#define TILE_BYTES (BM * LDT * 2)   // 10240B per buffered tile

// Issue one TDM load of a BMx32 bf16 tile (row-major, row pitch = strideElems)
// into LDS at byte offset ldsOff, with padding 4 DWORDs every 16 DWORDs
// (i.e. 64B row -> 80B LDS row == LDT bf16 elements).
__device__ __forceinline__ void tdm_load_tile(const __bf16* gaddr,
                                              unsigned tensorWidth,   // elements per row available
                                              unsigned rows,          // tile rows
                                              long strideElems,       // row pitch in elements
                                              unsigned ldsOff) {
  unsigned long long ga = (unsigned long long)(uintptr_t)gaddr;
  unsigned long long st = (unsigned long long)strideElems;
  u32x4 g0;
  g0[0] = 1u;                                             // count=1, user desc
  g0[1] = ldsOff;                                         // lds_addr
  g0[2] = (unsigned)(ga & 0xffffffffu);                   // global_addr[31:0]
  g0[3] = (unsigned)((ga >> 32) & 0x01ffffffu)            // global_addr[56:32]
        | 0x80000000u;                                    // type=2 ("image")
  i32x8 g1;
  g1[0] = (int)((1u << 16)       // data_size = 2 bytes
              | (1u << 20)       // pad_enable
              | (3u << 22)       // pad_interval: 16 DWORDs (64B) per chunk
              | (3u << 25));     // pad_amount: 4 DWORDs (16B) padding
  g1[1] = (int)((tensorWidth & 0xffffu) << 16);           // tensor_dim0 lo16
  g1[2] = (int)(((tensorWidth >> 16) & 0xffffu)           // tensor_dim0 hi16
              | ((rows & 0xffffu) << 16));                // tensor_dim1 lo16
  g1[3] = (int)(((rows >> 16) & 0xffffu)                  // tensor_dim1 hi16
              | ((unsigned)BK << 16));                    // tile_dim0 = 32
  g1[4] = (int)(rows & 0xffffu);                          // tile_dim1, tile_dim2=0
  g1[5] = (int)(unsigned)(st & 0xffffffffu);              // dim0_stride lo32
  g1[6] = (int)(unsigned)((st >> 32) & 0xffffu);          // dim0_stride hi16
  g1[7] = 0;
  i32x4 gz4 = {0, 0, 0, 0};                               // 2-D tensor: groups 2/3 null
  i32x8 gz8 = {0, 0, 0, 0, 0, 0, 0, 0};
  __builtin_amdgcn_tensor_load_to_lds(g0, g1, gz4, gz4, gz8, 0);
}

__global__ __launch_bounds__(256) void gemm_bf16_kernel(
    const __bf16* __restrict__ A, long lda, long aBatch,
    const __bf16* __restrict__ W, long ldb, long bBatch,
    float* __restrict__ C, __bf16* __restrict__ Cbf, long ldc, long cBatch,
    const float* __restrict__ bias, float alpha, int relu, int K)
{
  // Single LDS pool so byte offsets passed to the TDM descriptors are exact:
  // A ping @0, A pong @10240, B ping @20480, B pong @30720.
  __shared__ __align__(16) __bf16 lds_pool[4 * BM * LDT];

  const int tid  = threadIdx.x;
  const int lane = tid & 31;
  const int wid  = tid >> 5;
  const int wm   = wid & 3;     // 4 wave-rows  * 32 rows
  const int wn   = wid >> 2;    // 2 wave-cols  * 64 cols
  const int fr   = lane & 15;   // row within 16x16 fragment
  const int hh   = lane >> 4;   // K-half select

  const long m0 = (long)blockIdx.y * BM;
  const long n0 = (long)blockIdx.x * BN;
  const __bf16* Abase = A + (long)blockIdx.z * aBatch + m0 * lda;
  const __bf16* Bbase = W + (long)blockIdx.z * bBatch + n0 * ldb;
  if (C)   C   += (long)blockIdx.z * cBatch;
  if (Cbf) Cbf += (long)blockIdx.z * cBatch;

  v8f zero = {};
  v8f acc[2][4];
#pragma unroll
  for (int i = 0; i < 2; ++i)
#pragma unroll
    for (int j = 0; j < 4; ++j) acc[i][j] = zero;

  const int nk = K / BK;

  // Prologue: DMA first A/B tiles into ping buffers (wave 0 issues; TENSORcnt
  // is per-wave, so wave 0 waits and the workgroup barrier publishes LDS).
  if (wid == 0) {
    tdm_load_tile(Abase, (unsigned)K, BM, lda, 0u);
    tdm_load_tile(Bbase, (unsigned)K, BN, ldb, 2u * TILE_BYTES);
  }

  for (int i = 0; i < nk; ++i) {
    const int cur = i & 1;
    if (i + 1 < nk) {
      const int nxt = (i + 1) & 1;
      if (wid == 0) {
        tdm_load_tile(Abase + (long)(i + 1) * BK, (unsigned)K, BM, lda,
                      (unsigned)(nxt * TILE_BYTES));
        tdm_load_tile(Bbase + (long)(i + 1) * BK, (unsigned)K, BN, ldb,
                      (unsigned)(2 * TILE_BYTES + nxt * TILE_BYTES));
        __builtin_amdgcn_s_wait_tensorcnt(2);   // pair i landed, pair i+1 in flight
      }
    } else {
      if (wid == 0) __builtin_amdgcn_s_wait_tensorcnt(0);
    }
    __syncthreads();

    const __bf16* Abuf = lds_pool + cur * (BM * LDT);
    const __bf16* Bbuf = lds_pool + 2 * (BM * LDT) + cur * (BN * LDT);

    // Fragment gather: lane fr holds row (M or N) = fr; K chunks
    // {hh*8..hh*8+7, 16+hh*8..23+hh*8} per the 16-bit A-matrix layout.
    v16bf aF[2], bF[4];
#pragma unroll
    for (int ii = 0; ii < 2; ++ii) {
      int row = wm * 32 + ii * 16 + fr;
      v8bf lo = *(const v8bf*)&Abuf[row * LDT + hh * 8];
      v8bf hi = *(const v8bf*)&Abuf[row * LDT + 16 + hh * 8];
#pragma unroll
      for (int x = 0; x < 8; ++x) { aF[ii][x] = lo[x]; aF[ii][x + 8] = hi[x]; }
    }
#pragma unroll
    for (int j = 0; j < 4; ++j) {
      int row = wn * 64 + j * 16 + fr;
      v8bf lo = *(const v8bf*)&Bbuf[row * LDT + hh * 8];
      v8bf hi = *(const v8bf*)&Bbuf[row * LDT + 16 + hh * 8];
#pragma unroll
      for (int x = 0; x < 8; ++x) { bF[j][x] = lo[x]; bF[j][x + 8] = hi[x]; }
    }

#pragma unroll
    for (int ii = 0; ii < 2; ++ii)
#pragma unroll
      for (int j = 0; j < 4; ++j)
        acc[ii][j] = __builtin_amdgcn_wmma_f32_16x16x32_bf16(
            false, aF[ii], false, bF[j], (short)0, acc[ii][j], false, false);

    __syncthreads();   // all waves done with buf[cur] before TDM overwrites it
  }

  // Epilogue: C layout — lanes 0-15: N=lane, M=vr; lanes 16-31: N=lane-16, M=vr+8
#pragma unroll
  for (int i = 0; i < 2; ++i) {
#pragma unroll
    for (int j = 0; j < 4; ++j) {
      long col  = n0 + wn * 64 + j * 16 + fr;
      long rowb = m0 + wm * 32 + i * 16 + hh * 8;
      float bb = bias ? bias[col] : 0.0f;
#pragma unroll
      for (int vr = 0; vr < 8; ++vr) {
        float v = (acc[i][j][vr] + bb) * alpha;
        if (relu) v = fmaxf(v, 0.0f);
        long idx = (rowb + vr) * ldc + col;
        if (C)   C[idx]   = v;
        if (Cbf) Cbf[idx] = f2bf(v);
      }
    }
  }
}

// --------------------------- block reductions ------------------------------
__device__ __forceinline__ float blockSum(float v, float* red) {
  int tid = threadIdx.x;
  __syncthreads();
  red[tid] = v;
  __syncthreads();
  for (int off = 128; off > 0; off >>= 1) {
    if (tid < off) red[tid] += red[tid + off];
    __syncthreads();
  }
  float r = red[0];
  __syncthreads();
  return r;
}

__device__ __forceinline__ float blockMax(float v, float* red) {
  int tid = threadIdx.x;
  __syncthreads();
  red[tid] = v;
  __syncthreads();
  for (int off = 128; off > 0; off >>= 1) {
    if (tid < off) red[tid] = fmaxf(red[tid], red[tid + off]);
    __syncthreads();
  }
  float r = red[0];
  __syncthreads();
  return r;
}

// --------------- masked softmax over S for attention scores ----------------
// Block = b*T + t. mem_bias[b] is a per-(b) constant shift over s: it is
// softmax-invariant for kept lanes, so it is omitted.
__global__ __launch_bounds__(256) void attn_softmax_kernel(
    float* __restrict__ w, const unsigned char* __restrict__ mask)
{
  int bt = blockIdx.x;
  int b  = bt / T_;
  float* row = w + (long)bt * S_;
  __shared__ float red[256];
  int tid = threadIdx.x;
  bool k0 = mask[(long)tid * B_ + b] != 0;
  bool k1 = mask[(long)(tid + 256) * B_ + b] != 0;
  float v0 = k0 ? row[tid]       : -__builtin_inff();
  float v1 = k1 ? row[tid + 256] : -__builtin_inff();
  float mx = blockMax(fmaxf(v0, v1), red);
  float e0 = k0 ? __expf(v0 - mx) : 0.0f;
  float e1 = k1 ? __expf(v1 - mx) : 0.0f;
  float s  = blockSum(e0 + e1, red);
  float inv = 1.0f / s;
  row[tid]       = e0 * inv;
  row[tid + 256] = e1 * inv;
}

// ------------------------- attn = w @ v  (small) ---------------------------
__global__ __launch_bounds__(256) void attn_v_kernel(
    const float* __restrict__ w, const float* __restrict__ v,
    __bf16* __restrict__ out)
{
  int row = blockIdx.x;                 // t*B + b
  int t = row / B_, b = row % B_;
  __shared__ float ws[S_];
  int tid = threadIdx.x;
  const float* wrow = w + ((long)b * T_ + t) * S_;
  ws[tid]       = wrow[tid];
  ws[tid + 256] = wrow[tid + 256];
  __syncthreads();
  float acc[4] = {0.f, 0.f, 0.f, 0.f};
  for (int s = 0; s < S_; ++s) {
    float wv = ws[s];
    const float* vr = v + ((long)s * B_ + b) * E_;
#pragma unroll
    for (int j = 0; j < 4; ++j) acc[j] += wv * vr[tid + j * 256];
  }
#pragma unroll
  for (int j = 0; j < 4; ++j)
    out[(long)row * E_ + tid + j * 256] = f2bf(acc[j]);
}

// --------- fused: attn_norm LN, residual LN, divergence gates --------------
__global__ __launch_bounds__(256) void ln_gates_kernel(
    const float* __restrict__ outs, const float* __restrict__ attnproj,
    const float* __restrict__ g, const float* __restrict__ beta,
    const float* __restrict__ divw, const float* __restrict__ divb,
    __bf16* __restrict__ hln_bf, float* __restrict__ gates)
{
  int row = blockIdx.x;
  const float* o = outs     + (long)row * E_;
  const float* a = attnproj + (long)row * E_;
  int tid = threadIdx.x;
  __shared__ float red[256];

  float av[4], ov[4];
  float s1 = 0.f, s1q = 0.f, s2 = 0.f, s2q = 0.f;
#pragma unroll
  for (int j = 0; j < 4; ++j) {
    int e = tid + j * 256;
    av[j] = a[e]; ov[j] = o[e];
    s1 += av[j]; s1q += av[j] * av[j];
    float r = av[j] + ov[j];
    s2 += r; s2q += r * r;
  }
  s1  = blockSum(s1, red);  s1q = blockSum(s1q, red);
  s2  = blockSum(s2, red);  s2q = blockSum(s2q, red);
  const float invE = 1.0f / (float)E_;
  float m1 = s1 * invE, rs1 = rsqrtf(s1q * invE - m1 * m1 + LN_EPS);
  float m2 = s2 * invE, rs2 = rsqrtf(s2q * invE - m2 * m2 + LN_EPS);

  float d0 = 0.f, d1 = 0.f;
#pragma unroll
  for (int j = 0; j < 4; ++j) {
    int e = tid + j * 256;
    float ge = g[e], be = beta[e];
    float an = (av[j] - m1) * rs1 * ge + be;               // ln(attn)
    float hl = (av[j] + ov[j] - m2) * rs2 * ge + be;       // ln(outs+attn)
    hln_bf[(long)row * E_ + e] = f2bf(hl);
    d0 += ov[j] * divw[e]          + an * divw[E_ + e];
    d1 += ov[j] * divw[2 * E_ + e] + an * divw[3 * E_ + e];
  }
  d0 = blockSum(d0, red);
  d1 = blockSum(d1, red);
  if (tid == 0) {
    d0 += divb[0]; d1 += divb[1];
    float mx = fmaxf(d0, d1);
    float e0 = __expf(d0 - mx), e1 = __expf(d1 - mx);
    float inv = 1.0f / (e0 + e1);
    gates[row * 2 + 0] = e0 * inv;   // gen gate
    gates[row * 2 + 1] = e1 * inv;   // copy gate
  }
}

// ------------------------- plain LN -> bf16 --------------------------------
__global__ __launch_bounds__(256) void ln_to_bf16_kernel(
    const float* __restrict__ x, const float* __restrict__ g,
    const float* __restrict__ beta, __bf16* __restrict__ out)
{
  int row = blockIdx.x;
  const float* p = x + (long)row * E_;
  int tid = threadIdx.x;
  __shared__ float red[256];
  float xv[4];
  float s = 0.f, sq = 0.f;
#pragma unroll
  for (int j = 0; j < 4; ++j) {
    xv[j] = p[tid + j * 256];
    s += xv[j]; sq += xv[j] * xv[j];
  }
  s  = blockSum(s, red);
  sq = blockSum(sq, red);
  const float invE = 1.0f / (float)E_;
  float m = s * invE, rs = rsqrtf(sq * invE - m * m + LN_EPS);
#pragma unroll
  for (int j = 0; j < 4; ++j) {
    int e = tid + j * 256;
    out[(long)row * E_ + e] = f2bf((xv[j] - m) * rs * g[e] + beta[e]);
  }
}

// --------- in-place vocab softmax * gen_gate + copy-prob scatter -----------
__global__ __launch_bounds__(256) void vocab_softmax_copy_kernel(
    float* __restrict__ logits, const float* __restrict__ gates,
    const float* __restrict__ wprobs, const int* __restrict__ copy_seq)
{
  int row = blockIdx.x;                 // t*B + b
  int t = row / B_, b = row % B_;
  float* p = logits + (long)row * V_;
  int tid = threadIdx.x;
  __shared__ float red[256];

  float mx = -__builtin_inff();
  for (int i = tid; i < V_; i += 256) mx = fmaxf(mx, p[i]);
  mx = blockMax(mx, red);

  float sum = 0.f;
  for (int i = tid; i < V_; i += 256) sum += __expf(p[i] - mx);
  sum = blockSum(sum, red);

  float gen = gates[row * 2 + 0];
  float cpy = gates[row * 2 + 1];
  float scale = gen / sum;
  for (int i = tid; i < V_; i += 256) p[i] = __expf(p[i] - mx) * scale;
  __syncthreads();

  const float* arow = wprobs + ((long)b * T_ + t) * S_;   // alignment[t,b,:]
  for (int s = tid; s < S_; s += 256) {
    float c = cpy * arow[s];
    int idx = copy_seq[(long)s * B_ + b];
    atomicAdd(&p[idx], c);
  }
}

__global__ void log_kernel(float* __restrict__ p, long n) {
  long i = (long)blockIdx.x * blockDim.x + threadIdx.x;
  long stride = (long)gridDim.x * blockDim.x;
  for (; i < n; i += stride) p[i] = __logf(p[i] + 1e-12f);
}

// ---------------------------------------------------------------------------
extern "C" void kernel_launch(void* const* d_in, const int* in_sizes, int n_in,
                              void* d_out, int out_size, void* d_ws, size_t ws_size,
                              hipStream_t stream) {
  (void)in_sizes; (void)n_in; (void)out_size; (void)ws_size;
  const float* outs       = (const float*)d_in[0];
  const float* mem        = (const float*)d_in[1];
  const float* mem_bias   = (const float*)d_in[2]; (void)mem_bias; // softmax-invariant shift
  const float* in_proj_w  = (const float*)d_in[3];
  const float* in_proj_b  = (const float*)d_in[4];
  const float* out_proj_w = (const float*)d_in[5];
  const float* out_proj_b = (const float*)d_in[6];
  const float* aln_g      = (const float*)d_in[7];
  const float* aln_b      = (const float*)d_in[8];
  const float* div_w      = (const float*)d_in[9];
  const float* div_b      = (const float*)d_in[10];
  const float* fc1_w      = (const float*)d_in[11];
  const float* fc1_b      = (const float*)d_in[12];
  const float* fc2_w      = (const float*)d_in[13];
  const float* fc2_b      = (const float*)d_in[14];
  const float* ffn_g      = (const float*)d_in[15];
  const float* ffn_b      = (const float*)d_in[16];
  const float* vocab_w    = (const float*)d_in[17];
  const unsigned char* mem_mask = (const unsigned char*)d_in[18];
  const int* copy_seq     = (const int*)d_in[19];
  float* out = (float*)d_out;

  char* ws = (char*)d_ws;
  size_t off = 0;
  auto alloc = [&](size_t bytes) -> void* {
    off = (off + 255) & ~(size_t)255;
    void* p = ws + off;
    off += bytes;
    return p;
  };

  __bf16* vocab_bf   = (__bf16*)alloc((size_t)V_ * E_ * 2);
  __bf16* inproj_bf  = (__bf16*)alloc((size_t)3 * E_ * E_ * 2);
  __bf16* outproj_bf = (__bf16*)alloc((size_t)E_ * E_ * 2);
  __bf16* fc1_bf     = (__bf16*)alloc((size_t)FF_ * E_ * 2);
  __bf16* fc2_bf     = (__bf16*)alloc((size_t)E_ * FF_ * 2);
  __bf16* outs_bf    = (__bf16*)alloc((size_t)T_ * B_ * E_ * 2);
  __bf16* mem_bf     = (__bf16*)alloc((size_t)S_ * B_ * E_ * 2);
  __bf16* q_bf       = (__bf16*)alloc((size_t)T_ * B_ * E_ * 2);
  __bf16* k_bf       = (__bf16*)alloc((size_t)S_ * B_ * E_ * 2);
  float*  v_f32      = (float*) alloc((size_t)S_ * B_ * E_ * 4);
  float*  wraw       = (float*) alloc((size_t)B_ * T_ * S_ * 4);
  __bf16* attn_bf    = (__bf16*)alloc((size_t)T_ * B_ * E_ * 2);
  float*  attnproj   = (float*) alloc((size_t)T_ * B_ * E_ * 4);
  __bf16* hln_bf     = (__bf16*)alloc((size_t)T_ * B_ * E_ * 2);
  float*  gates      = (float*) alloc((size_t)T_ * B_ * 2 * 4);
  __bf16* ff_bf      = (__bf16*)alloc((size_t)T_ * B_ * FF_ * 2);
  float*  h_f32      = (float*) alloc((size_t)T_ * B_ * E_ * 4);
  __bf16* h2_bf      = (__bf16*)alloc((size_t)T_ * B_ * E_ * 2);

  // --- stage 0: fp32 -> bf16 conversions (weights + activations) ---
  cvt_bf16_kernel<<<8192, 256, 0, stream>>>(vocab_w,    vocab_bf,   (long)V_ * E_);
  cvt_bf16_kernel<<<4096, 256, 0, stream>>>(in_proj_w,  inproj_bf,  (long)3 * E_ * E_);
  cvt_bf16_kernel<<<2048, 256, 0, stream>>>(out_proj_w, outproj_bf, (long)E_ * E_);
  cvt_bf16_kernel<<<4096, 256, 0, stream>>>(fc1_w,      fc1_bf,     (long)FF_ * E_);
  cvt_bf16_kernel<<<4096, 256, 0, stream>>>(fc2_w,      fc2_bf,     (long)E_ * FF_);
  cvt_bf16_kernel<<<2048, 256, 0, stream>>>(outs,       outs_bf,    (long)T_ * B_ * E_);
  cvt_bf16_kernel<<<4096, 256, 0, stream>>>(mem,        mem_bf,     (long)S_ * B_ * E_);

  // --- stage 1: q/k/v projections (q scaled by E^-0.5 = 1/32) ---
  gemm_bf16_kernel<<<dim3(E_ / BN, (T_ * B_) / BM, 1), 256, 0, stream>>>(
      outs_bf, E_, 0, inproj_bf, E_, 0,
      nullptr, q_bf, E_, 0, in_proj_b, 0.03125f, 0, E_);
  gemm_bf16_kernel<<<dim3(E_ / BN, (S_ * B_) / BM, 1), 256, 0, stream>>>(
      mem_bf, E_, 0, inproj_bf + (size_t)E_ * E_, E_, 0,
      nullptr, k_bf, E_, 0, in_proj_b + E_, 1.0f, 0, E_);
  gemm_bf16_kernel<<<dim3(E_ / BN, (S_ * B_) / BM, 1), 256, 0, stream>>>(
      mem_bf, E_, 0, inproj_bf + (size_t)2 * E_ * E_, E_, 0,
      v_f32, nullptr, E_, 0, in_proj_b + 2 * E_, 1.0f, 0, E_);

  // --- stage 2: scores w[b,t,s] = q_b @ k_b^T (batched WMMA over b) ---
  gemm_bf16_kernel<<<dim3(S_ / BN, T_ / BM, B_), 256, 0, stream>>>(
      q_bf, (long)B_ * E_, E_, k_bf, (long)B_ * E_, E_,
      wraw, nullptr, S_, (long)T_ * S_, nullptr, 1.0f, 0, E_);

  attn_softmax_kernel<<<B_ * T_, 256, 0, stream>>>(wraw, mem_mask);
  attn_v_kernel<<<T_ * B_, 256, 0, stream>>>(wraw, v_f32, attn_bf);

  // --- stage 3: out_proj + fused LN/gates ---
  gemm_bf16_kernel<<<dim3(E_ / BN, (T_ * B_) / BM, 1), 256, 0, stream>>>(
      attn_bf, E_, 0, outproj_bf, E_, 0,
      attnproj, nullptr, E_, 0, out_proj_b, 1.0f, 0, E_);
  ln_gates_kernel<<<T_ * B_, 256, 0, stream>>>(
      outs, attnproj, aln_g, aln_b, div_w, div_b, hln_bf, gates);

  // --- stage 4: FFN ---
  gemm_bf16_kernel<<<dim3(FF_ / BN, (T_ * B_) / BM, 1), 256, 0, stream>>>(
      hln_bf, E_, 0, fc1_bf, E_, 0,
      nullptr, ff_bf, FF_, 0, fc1_b, 1.0f, 1, E_);
  gemm_bf16_kernel<<<dim3(E_ / BN, (T_ * B_) / BM, 1), 256, 0, stream>>>(
      ff_bf, FF_, 0, fc2_bf, FF_, 0,
      h_f32, nullptr, E_, 0, fc2_b, 1.0f, 0, FF_);
  ln_to_bf16_kernel<<<T_ * B_, 256, 0, stream>>>(h_f32, ffn_g, ffn_b, h2_bf);

  // --- stage 5: vocab logits straight into d_out (dominant GEMM, 134 GF) ---
  gemm_bf16_kernel<<<dim3(V_ / BN, (T_ * B_) / BM, 1), 256, 0, stream>>>(
      h2_bf, E_, 0, vocab_bf, E_, 0,
      out, nullptr, V_, 0, nullptr, 1.0f, 0, E_);

  // --- stage 6: softmax * gen_gate + copy scatter, then log ---
  vocab_softmax_copy_kernel<<<T_ * B_, 256, 0, stream>>>(out, gates, wraw, copy_seq);
  log_kernel<<<4096, 256, 0, stream>>>(out, (long)T_ * B_ * V_);
}
